// MareyTransformer_15212774162825
// MI455X (gfx1250) — compile-verified, matrix operand order checked by reading, not run on hardware
//
#include <hip/hip_runtime.h>

// ---------------------------------------------------------------- constants
#define Dm     1536
#define Hh     16
#define HDm    96
#define ROPEC  48
#define FFm    6144
#define NXm    2048
#define NYm    256
#define NTm    2304
#define EPSf   1e-6f

// ---------------------------------------------------------------- types
typedef __attribute__((ext_vector_type(16))) __bf16 v16bf;
typedef __attribute__((ext_vector_type(8)))  __bf16 v8bf;
typedef __attribute__((ext_vector_type(4)))  __bf16 v4bf;
typedef __attribute__((ext_vector_type(8)))  float  v8f;

// Native fp32 -> bf16 (fptrunc, RNE) — lowers to hardware cvt on gfx1250.
__device__ __forceinline__ __bf16 f2bf(float f) { return (__bf16)f; }

__device__ __forceinline__ v8f wmma_bf16(v16bf a, v16bf b, v8f c) {
  // D = A(16x32) * B(32x16) + C, fp32 accumulate
  return __builtin_amdgcn_wmma_f32_16x16x32_bf16(
      false, a, false, b, (short)0, c, false, false);
}

// Build a 16-bf16 A/B fragment from two contiguous 8-element halves
// (per ISA 16-bit operand layout: lane holds k = kb..kb+7 and kb+16..kb+23).
__device__ __forceinline__ v16bf frag16(const __bf16* p0, const __bf16* p1) {
  v8bf lo = *(const v8bf*)p0;
  v8bf hi = *(const v8bf*)p1;
  return __builtin_shufflevector(lo, hi, 0, 1, 2, 3, 4, 5, 6, 7,
                                         8, 9, 10, 11, 12, 13, 14, 15);
}

__device__ __forceinline__ float wave_sum(float v) {
#pragma unroll
  for (int m = 16; m >= 1; m >>= 1) v += __shfl_xor(v, m, 32);
  return v;
}

// ================================================================ kernel 1:
// st = silu(t);  mod_{x,y} = st @ w_mod_{x,y} + b   (1 x 1536 @ 1536 x 9216)
__global__ __launch_bounds__(256)
void k_modvec(const float* __restrict__ t,
              const float* __restrict__ wx, const float* __restrict__ bx,
              const float* __restrict__ wy, const float* __restrict__ by,
              float* __restrict__ mx, float* __restrict__ my) {
  __shared__ float st[Dm];
  const int tid = threadIdx.x;
  for (int j = tid; j < Dm; j += 256) {
    float v = t[j];
    st[j] = v / (1.f + __expf(-v));
  }
  __syncthreads();
  const int mat = blockIdx.x / 36;
  const int col = (blockIdx.x % 36) * 256 + tid;
  const float* W = mat ? wy : wx;
  const float* B = mat ? by : bx;
  float* O = mat ? my : mx;
  float acc = 0.f;
  for (int k = 0; k < Dm; ++k) acc += st[k] * W[(size_t)k * (6 * Dm) + col];
  O[col] = acc + B[col];
}

// ================================================================ kernel 2:
// LayerNorm + AdaLN modulation -> bf16 activations (one block per token row)
__global__ __launch_bounds__(256)
void k_ln_mod(const float* __restrict__ A, const float* __restrict__ B,
              int rowSplit,
              const float* __restrict__ modx, const float* __restrict__ mody,
              int shOff, int scOff, __bf16* __restrict__ out) {
  __shared__ float rs[8], rs2[8];
  const int row = blockIdx.x, tid = threadIdx.x;
  const float* src = (row < rowSplit) ? (A + (size_t)row * Dm)
                                      : (B + (size_t)(row - rowSplit) * Dm);
  const float* mod = (row < rowSplit) ? modx : mody;
  float xl[6], s = 0.f, s2 = 0.f;
#pragma unroll
  for (int i = 0; i < 6; ++i) {
    float v = src[tid + i * 256];
    xl[i] = v; s += v; s2 += v * v;
  }
  s = wave_sum(s); s2 = wave_sum(s2);
  if ((tid & 31) == 0) { rs[tid >> 5] = s; rs2[tid >> 5] = s2; }
  __syncthreads();
  s = 0.f; s2 = 0.f;
#pragma unroll
  for (int w = 0; w < 8; ++w) { s += rs[w]; s2 += rs2[w]; }
  const float mean = s * (1.f / Dm);
  const float var  = s2 * (1.f / Dm) - mean * mean;
  const float inv  = rsqrtf(var + EPSf);
#pragma unroll
  for (int i = 0; i < 6; ++i) {
    const int j = tid + i * 256;
    const float xh = (xl[i] - mean) * inv;
    out[(size_t)row * Dm + j] = f2bf(xh * (1.f + mod[scOff + j]) + mod[shOff + j]);
  }
}

// ================================================================ kernel 3:
// Generic bf16 WMMA GEMM: C[M,N] = A[M,K](bf16) @ W[K,N](fp32, cvt on load)
// EPI: 1 = acc+bias | 2 = res + gate*(acc+bias) | 3 = res + gate*acc
#define BM 64
#define BN 128
#define BK 32
template <int EPI>
__global__ __launch_bounds__(256)
void k_gemm(const __bf16* __restrict__ A, const float* __restrict__ W,
            const float* __restrict__ bias, const float* __restrict__ gate,
            const float* __restrict__ res, float* __restrict__ C,
            int M, int N, int K) {
  __shared__ __bf16 As[BM][BK];
  __shared__ __bf16 Bs[BN][BK];   // transposed: [n][k]

  const int tid  = threadIdx.x;
  const int lane = tid & 31;
  const int wave = tid >> 5;
  const int n0 = blockIdx.x * BN;
  const int m0 = blockIdx.y * BM;
  const int wm = (wave >> 2) * 32;
  const int wn = (wave & 3) * 32;

  const int ar  = tid >> 2;          // A-tile row 0..63
  const int ac  = (tid & 3) * 8;     // A-tile col chunk
  const int bn  = tid & 127;         // B-tile n
  const int bk0 = (tid >> 7) * 4;    // B-tile k sub-chunk
  const int lm  = lane & 15;
  const int kb  = (lane < 16) ? 0 : 8;
  const int rhalf = (lane < 16) ? 0 : 8;

  v8f c00 = {}, c01 = {}, c10 = {}, c11 = {};

  for (int kt = 0; kt < K; kt += BK) {
    // stage A (already bf16, 16B vector copies)
    *(v8bf*)&As[ar][ac] = *(const v8bf*)&A[(size_t)(m0 + ar) * K + kt + ac];
    // stage B: fp32 -> bf16 (hardware cvt), transposed to [n][k]
#pragma unroll
    for (int p = 0; p < 4; ++p) {
      const int k = p * 8 + bk0;
      v4bf pk;
#pragma unroll
      for (int i = 0; i < 4; ++i)
        pk[i] = f2bf(W[(size_t)(kt + k + i) * N + n0 + bn]);
      *(v4bf*)&Bs[bn][k] = pk;
    }
    if (kt + BK < K)
      __builtin_prefetch(&W[(size_t)(kt + BK + bk0) * N + n0 + bn], 0, 1);
    __syncthreads();

    v16bf a0 = frag16(&As[wm + lm][kb],      &As[wm + lm][kb + 16]);
    v16bf a1 = frag16(&As[wm + 16 + lm][kb], &As[wm + 16 + lm][kb + 16]);
    v16bf b0 = frag16(&Bs[wn + lm][kb],      &Bs[wn + lm][kb + 16]);
    v16bf b1 = frag16(&Bs[wn + 16 + lm][kb], &Bs[wn + 16 + lm][kb + 16]);
    c00 = wmma_bf16(a0, b0, c00);
    c01 = wmma_bf16(a0, b1, c01);
    c10 = wmma_bf16(a1, b0, c10);
    c11 = wmma_bf16(a1, b1, c11);
    __syncthreads();
  }

  auto epi = [&](v8f acc, int mb, int nb) {
    const int col = nb + lm;
    const int rb  = mb + rhalf;
    float bi = 0.f, g = 1.f;
    if constexpr (EPI == 1 || EPI == 2) bi = bias[col];
    if constexpr (EPI >= 2)             g  = gate[col];
#pragma unroll
    for (int r = 0; r < 8; ++r) {
      float v = acc[r] + bi;
      if constexpr (EPI >= 2) v = res[(size_t)(rb + r) * N + col] + g * v;
      C[(size_t)(rb + r) * N + col] = v;
    }
  };
  epi(c00, m0 + wm,      n0 + wn);
  epi(c01, m0 + wm,      n0 + wn + 16);
  epi(c10, m0 + wm + 16, n0 + wn);
  epi(c11, m0 + wm + 16, n0 + wn + 16);
}

// ================================================================ kernel 3b:
// Fused SwiGLU GEMM: H = silu(A@W1) * (A@W2), bf16 output
__global__ __launch_bounds__(256)
void k_gemm_swiglu(const __bf16* __restrict__ A, const float* __restrict__ W1,
                   const float* __restrict__ W2, __bf16* __restrict__ Hout,
                   int M, int N, int K) {
  __shared__ __bf16 As[BM][BK];
  __shared__ __bf16 B1s[BN][BK];
  __shared__ __bf16 B2s[BN][BK];

  const int tid  = threadIdx.x;
  const int lane = tid & 31;
  const int wave = tid >> 5;
  const int n0 = blockIdx.x * BN;
  const int m0 = blockIdx.y * BM;
  const int wm = (wave >> 2) * 32;
  const int wn = (wave & 3) * 32;
  const int ar  = tid >> 2;
  const int ac  = (tid & 3) * 8;
  const int bn  = tid & 127;
  const int bk0 = (tid >> 7) * 4;
  const int lm  = lane & 15;
  const int kb  = (lane < 16) ? 0 : 8;
  const int rhalf = (lane < 16) ? 0 : 8;

  v8f c1[4] = {}, c2[4] = {};

  for (int kt = 0; kt < K; kt += BK) {
    *(v8bf*)&As[ar][ac] = *(const v8bf*)&A[(size_t)(m0 + ar) * K + kt + ac];
#pragma unroll
    for (int p = 0; p < 4; ++p) {
      const int k = p * 8 + bk0;
      v4bf p1, p2;
#pragma unroll
      for (int i = 0; i < 4; ++i) {
        p1[i] = f2bf(W1[(size_t)(kt + k + i) * N + n0 + bn]);
        p2[i] = f2bf(W2[(size_t)(kt + k + i) * N + n0 + bn]);
      }
      *(v4bf*)&B1s[bn][k] = p1;
      *(v4bf*)&B2s[bn][k] = p2;
    }
    __syncthreads();

    v16bf a0 = frag16(&As[wm + lm][kb],      &As[wm + lm][kb + 16]);
    v16bf a1 = frag16(&As[wm + 16 + lm][kb], &As[wm + 16 + lm][kb + 16]);
    v16bf u0 = frag16(&B1s[wn + lm][kb],      &B1s[wn + lm][kb + 16]);
    v16bf u1 = frag16(&B1s[wn + 16 + lm][kb], &B1s[wn + 16 + lm][kb + 16]);
    v16bf w0 = frag16(&B2s[wn + lm][kb],      &B2s[wn + lm][kb + 16]);
    v16bf w1 = frag16(&B2s[wn + 16 + lm][kb], &B2s[wn + 16 + lm][kb + 16]);
    c1[0] = wmma_bf16(a0, u0, c1[0]);  c1[1] = wmma_bf16(a0, u1, c1[1]);
    c1[2] = wmma_bf16(a1, u0, c1[2]);  c1[3] = wmma_bf16(a1, u1, c1[3]);
    c2[0] = wmma_bf16(a0, w0, c2[0]);  c2[1] = wmma_bf16(a0, w1, c2[1]);
    c2[2] = wmma_bf16(a1, w0, c2[2]);  c2[3] = wmma_bf16(a1, w1, c2[3]);
    __syncthreads();
  }

#pragma unroll
  for (int t = 0; t < 4; ++t) {
    const int mb = m0 + wm + (t >> 1) * 16;
    const int nb = n0 + wn + (t & 1) * 16;
    const int col = nb + lm;
    const int rb  = mb + rhalf;
#pragma unroll
    for (int r = 0; r < 8; ++r) {
      const float a = c1[t][r];
      const float h = (a / (1.f + __expf(-a))) * c2[t][r];
      Hout[(size_t)(rb + r) * N + col] = f2bf(h);
    }
  }
}

// ================================================================ kernel 4:
// Per (token, head): RMS-norm q/k, RoPE (x-stream), scatter to
// Q[h][tok][d], K[h][tok][d], Vt[h][d][tok] bf16.  1/sqrt(HD) folded into Q.
__global__ __launch_bounds__(96)
void k_qkv_post(const float* __restrict__ qkv, const int* __restrict__ positions,
                const float* __restrict__ qnx, const float* __restrict__ knx,
                const float* __restrict__ qny, const float* __restrict__ kny,
                __bf16* __restrict__ Qo, __bf16* __restrict__ Ko,
                __bf16* __restrict__ Vto) {
  __shared__ float red[6];
  __shared__ float sq[HDm], sk[HDm];
  const int token = blockIdx.x;
  const int h     = blockIdx.y;
  const int d     = threadIdx.x;
  const int lane  = d & 31, wave = d >> 5;
  const bool isX  = token < NXm;

  const float* base = qkv + (size_t)token * (3 * Dm);
  const float qv = base[h * HDm + d];
  const float kv = base[Dm + h * HDm + d];
  const float vv = base[2 * Dm + h * HDm + d];

  float q2 = wave_sum(qv * qv);
  float k2 = wave_sum(kv * kv);
  if (lane == 0) { red[wave] = q2; red[3 + wave] = k2; }
  __syncthreads();
  const float qr = rsqrtf((red[0] + red[1] + red[2]) * (1.f / HDm) + EPSf);
  const float kr = rsqrtf((red[3] + red[4] + red[5]) * (1.f / HDm) + EPSf);

  const float* qn = isX ? qnx : qny;
  const float* kn = isX ? knx : kny;
  float qv2 = qv * qr * qn[d];
  float kv2 = kv * kr * kn[d];
  sq[d] = qv2; sk[d] = kv2;
  __syncthreads();

  if (isX && d >= ROPEC) {
    const float pos = (float)positions[token];
    const int i = (d < ROPEC + 24) ? (d - ROPEC) : (d - ROPEC - 24);
    const float ts = __powf(10000.f, 2.f * (float)i / (float)ROPEC);
    float s, c; __sincosf(pos / ts, &s, &c);
    const float aq = sq[ROPEC + i], bq = sq[ROPEC + 24 + i];
    const float ak = sk[ROPEC + i], bk = sk[ROPEC + 24 + i];
    if (d < ROPEC + 24) { qv2 = aq * c - bq * s; kv2 = ak * c - bk * s; }
    else                { qv2 = bq * c + aq * s; kv2 = bk * c + ak * s; }
  }
  const float qscale = rsqrtf((float)HDm);
  Qo[((size_t)h * NTm + token) * HDm + d]  = f2bf(qv2 * qscale);
  Ko[((size_t)h * NTm + token) * HDm + d]  = f2bf(kv2);
  Vto[((size_t)h * HDm + d) * NTm + token] = f2bf(vv);
}

// ================================================================ kernel 5:
// Flash attention: each wave handles 16 query rows for one head;
// keys processed in chunks of 32; WMMA for QK^T and P@V.
__global__ __launch_bounds__(256)
void k_attn(const __bf16* __restrict__ Q, const __bf16* __restrict__ Kb,
            const __bf16* __restrict__ Vt, __bf16* __restrict__ Obf) {
  // per-wave P staging, bf16 (converted once on write); row stride 40
  // keeps the v8bf fragment reads 16B-aligned.
  __shared__ __bf16 sPb[8][16][40];
  const int lane = threadIdx.x & 31;
  const int wave = threadIdx.x >> 5;
  const int head = blockIdx.y;
  const int q0   = (blockIdx.x * 8 + wave) * 16;
  const __bf16* Qh = Q  + (size_t)head * NTm * HDm;
  const __bf16* Kh = Kb + (size_t)head * NTm * HDm;
  const __bf16* Vh = Vt + (size_t)head * HDm * NTm;
  const int lm = lane & 15;
  const int kb = (lane < 16) ? 0 : 8;
  const int rhalf = (lane < 16) ? 0 : 8;

  v16bf qf[3];
#pragma unroll
  for (int t = 0; t < 3; ++t) {
    const __bf16* p = Qh + (size_t)(q0 + lm) * HDm + t * 32 + kb;
    qf[t] = frag16(p, p + 16);
  }

  v8f o[6] = {};
  float mrow[8], lrow[8];
#pragma unroll
  for (int r = 0; r < 8; ++r) { mrow[r] = -3.0e38f; lrow[r] = 0.f; }

  for (int kc = 0; kc < NTm; kc += 32) {
    v8f s0 = {}, s1 = {};
#pragma unroll
    for (int t = 0; t < 3; ++t) {
      const __bf16* p0 = Kh + (size_t)(kc + lm) * HDm + t * 32 + kb;
      const __bf16* p1 = Kh + (size_t)(kc + 16 + lm) * HDm + t * 32 + kb;
      s0 = wmma_bf16(qf[t], frag16(p0, p0 + 16), s0);
      s1 = wmma_bf16(qf[t], frag16(p1, p1 + 16), s1);
    }
    // online softmax (rows live in 16-lane halves; reduce across cols)
#pragma unroll
    for (int r = 0; r < 8; ++r) {
      float mx = fmaxf(s0[r], s1[r]);
      mx = fmaxf(mx, __shfl_xor(mx, 1, 32));
      mx = fmaxf(mx, __shfl_xor(mx, 2, 32));
      mx = fmaxf(mx, __shfl_xor(mx, 4, 32));
      mx = fmaxf(mx, __shfl_xor(mx, 8, 32));
      const float mn = fmaxf(mrow[r], mx);
      const float corr = __expf(mrow[r] - mn);
      mrow[r] = mn;
      const float p0 = __expf(s0[r] - mn);
      const float p1 = __expf(s1[r] - mn);
      float ps = p0 + p1;
      ps += __shfl_xor(ps, 1, 32);
      ps += __shfl_xor(ps, 2, 32);
      ps += __shfl_xor(ps, 4, 32);
      ps += __shfl_xor(ps, 8, 32);
      lrow[r] = lrow[r] * corr + ps;
#pragma unroll
      for (int c = 0; c < 6; ++c) o[c][r] *= corr;
      sPb[wave][rhalf + r][lm]      = f2bf(p0);
      sPb[wave][rhalf + r][lm + 16] = f2bf(p1);
    }
    // transpose P (C-layout -> A-fragment) through LDS, two b128 reads
    v16bf pf = frag16(&sPb[wave][lm][kb], &sPb[wave][lm][kb + 16]);
    // O += P @ V
#pragma unroll
    for (int c = 0; c < 6; ++c) {
      const __bf16* vp = Vh + (size_t)(c * 16 + lm) * NTm + kc + kb;
      o[c] = wmma_bf16(pf, frag16(vp, vp + 16), o[c]);
    }
  }

  float invl[8];
#pragma unroll
  for (int r = 0; r < 8; ++r) invl[r] = 1.f / lrow[r];
#pragma unroll
  for (int c = 0; c < 6; ++c)
#pragma unroll
    for (int r = 0; r < 8; ++r) {
      const int token = q0 + rhalf + r;
      Obf[(size_t)token * Dm + head * HDm + c * 16 + lm] = f2bf(o[c][r] * invl[r]);
    }
}

// ================================================================ host
extern "C" void kernel_launch(void* const* d_in, const int* in_sizes, int n_in,
                              void* d_out, int out_size, void* d_ws, size_t ws_size,
                              hipStream_t stream) {
  (void)in_sizes; (void)n_in; (void)out_size; (void)ws_size;
  const float* x        = (const float*)d_in[0];
  const float* y        = (const float*)d_in[1];
  const float* t        = (const float*)d_in[2];
  const int*   positions= (const int*)  d_in[3];
  const float* w_mod_x  = (const float*)d_in[4];
  const float* b_mod_x  = (const float*)d_in[5];
  const float* w_mod_y  = (const float*)d_in[6];
  const float* b_mod_y  = (const float*)d_in[7];
  const float* w_qkv_x  = (const float*)d_in[8];
  const float* b_qkv_x  = (const float*)d_in[9];
  const float* w_qkv_y  = (const float*)d_in[10];
  const float* b_qkv_y  = (const float*)d_in[11];
  const float* qn_x     = (const float*)d_in[12];
  const float* kn_x     = (const float*)d_in[13];
  const float* qn_y     = (const float*)d_in[14];
  const float* kn_y     = (const float*)d_in[15];
  const float* w_proj_x = (const float*)d_in[16];
  const float* b_proj_x = (const float*)d_in[17];
  const float* w_proj_y = (const float*)d_in[18];
  const float* b_proj_y = (const float*)d_in[19];
  const float* w1_x     = (const float*)d_in[20];
  const float* w2_x     = (const float*)d_in[21];
  const float* w3_x     = (const float*)d_in[22];
  const float* w1_y     = (const float*)d_in[23];
  const float* w2_y     = (const float*)d_in[24];
  const float* w3_y     = (const float*)d_in[25];
  float* out = (float*)d_out;

  // ---- workspace carve-up (~120 MB)
  char* w = (char*)d_ws;
  float*  mx  = (float*)w;               w += (size_t)6 * Dm * 4;
  float*  my  = (float*)w;               w += (size_t)6 * Dm * 4;
  __bf16* zbf = (__bf16*)w;              w += (size_t)NTm * Dm * 2;
  float*  qkv = (float*)w;               w += (size_t)NTm * 3 * Dm * 4;
  __bf16* qh  = (__bf16*)w;              w += (size_t)Hh * NTm * HDm * 2;
  __bf16* kh  = (__bf16*)w;              w += (size_t)Hh * NTm * HDm * 2;
  __bf16* vth = (__bf16*)w;              w += (size_t)Hh * HDm * NTm * 2;
  __bf16* obf = (__bf16*)w;              w += (size_t)NTm * Dm * 2;
  float*  x1  = (float*)w;               w += (size_t)NTm * Dm * 4;
  __bf16* hbf = (__bf16*)w;              w += (size_t)NTm * FFm * 2;

  const dim3 blk(256);

  // 1) modulation vectors
  k_modvec<<<72, blk, 0, stream>>>(t, w_mod_x, b_mod_x, w_mod_y, b_mod_y, mx, my);
  // 2) LN1 + mod -> bf16
  k_ln_mod<<<NTm, blk, 0, stream>>>(x, y, NXm, mx, my, 0, Dm, zbf);
  // 3) QKV GEMMs (bias)
  k_gemm<1><<<dim3(3 * Dm / BN, NXm / BM), blk, 0, stream>>>(
      zbf, w_qkv_x, b_qkv_x, nullptr, nullptr, qkv, NXm, 3 * Dm, Dm);
  k_gemm<1><<<dim3(3 * Dm / BN, NYm / BM), blk, 0, stream>>>(
      zbf + (size_t)NXm * Dm, w_qkv_y, b_qkv_y, nullptr, nullptr,
      qkv + (size_t)NXm * 3 * Dm, NYm, 3 * Dm, Dm);
  // 4) RMS + RoPE + scatter
  k_qkv_post<<<dim3(NTm, Hh), 96, 0, stream>>>(qkv, positions,
      qn_x, kn_x, qn_y, kn_y, qh, kh, vth);
  // 5) attention
  k_attn<<<dim3(NTm / (16 * 8), Hh), blk, 0, stream>>>(qh, kh, vth, obf);
  // 6) out-proj + gate + residual -> x1
  k_gemm<2><<<dim3(Dm / BN, NXm / BM), blk, 0, stream>>>(
      obf, w_proj_x, b_proj_x, mx + 2 * Dm, x, x1, NXm, Dm, Dm);
  k_gemm<2><<<dim3(Dm / BN, NYm / BM), blk, 0, stream>>>(
      obf + (size_t)NXm * Dm, w_proj_y, b_proj_y, my + 2 * Dm, y,
      x1 + (size_t)NXm * Dm, NYm, Dm, Dm);
  // 7) LN2 + mod -> bf16 (reuse zbf)
  k_ln_mod<<<NTm, blk, 0, stream>>>(x1, x1 + (size_t)NXm * Dm, NXm,
                                    mx, my, 3 * Dm, 4 * Dm, zbf);
  // 8) fused SwiGLU GEMM -> hbf (bf16)
  k_gemm_swiglu<<<dim3(FFm / BN, NXm / BM), blk, 0, stream>>>(
      zbf, w1_x, w2_x, hbf, NXm, FFm, Dm);
  k_gemm_swiglu<<<dim3(FFm / BN, NYm / BM), blk, 0, stream>>>(
      zbf + (size_t)NXm * Dm, w1_y, w2_y, hbf + (size_t)NXm * FFm, NYm, FFm, Dm);
  // 9) down-proj + gate + residual -> d_out
  k_gemm<3><<<dim3(Dm / BN, NXm / BM), blk, 0, stream>>>(
      hbf, w3_x, nullptr, mx + 5 * Dm, x1, out, NXm, Dm, FFm);
  k_gemm<3><<<dim3(Dm / BN, NYm / BM), blk, 0, stream>>>(
      hbf + (size_t)NXm * FFm, w3_y, nullptr, my + 5 * Dm,
      x1 + (size_t)NXm * Dm, out + (size_t)NXm * Dm, NYm, Dm, FFm);
}
